// CrossAttentionWithMask_2688649527353
// MI455X (gfx1250) — matrix operand order repaired and checked by reading, not compiled, not run on hardware
//
#include <hip/hip_runtime.h>

// ---------------------------------------------------------------------------
// CrossAttentionWithMask for MI455X (gfx1250, wave32, WMMA).
//   B=1, C=64, H=W=64 -> L=4096, N=4.
//   Attention ~17.2 GFLOP, all operands L2-resident (~9 MB) -> matrix-pipe
//   bound. v_wmma_f32_16x16x32_bf16 for QK^T and PV. Projections (~0.3 GFLOP)
//   on VALU fp32. max_attn == 1/rowsum falls out of online softmax.
//   v2: software-pipelined K (double-buffered regs) + early V loads.
//   v3: softmax row-sums via WMMA against an all-ones B fragment (P·1), which
//       removes 32 ds_bpermute per step and normalizes by the same bf16 P
//       that the PV product consumes.
// ---------------------------------------------------------------------------

#define CC 64
#define LL 4096
#define NN 4

typedef __attribute__((ext_vector_type(16))) __bf16 v16bf;
typedef __attribute__((ext_vector_type(8)))  float  v8f;

union FragU { struct { uint4 lo, hi; } u; v16bf v; };

static __device__ __forceinline__ v16bf load_frag(const unsigned short* p0,
                                                  const unsigned short* p1) {
  FragU f;
  f.u.lo = *(const uint4*)p0;   // elements 0..7
  f.u.hi = *(const uint4*)p1;   // elements 8..15
  return f.v;
}

static __device__ __forceinline__ v16bf ones_frag() {
  FragU f;
  const unsigned one2 = 0x3F803F80u;          // two bf16 1.0
  f.u.lo = (uint4){one2, one2, one2, one2};
  f.u.hi = (uint4){one2, one2, one2, one2};
  return f.v;
}

static __device__ __forceinline__ unsigned short f2bf(float f) {
  unsigned u = __float_as_uint(f);
  u += 0x7FFFu + ((u >> 16) & 1u);            // round-to-nearest-even
  return (unsigned short)(u >> 16);
}

// ---------------------------------------------------------------------------
// Projection kernel: 9 tasks x 64 row-groups.
//   task 0      : Q  = (query^T @ Wq^T + bq) * (1/sqrt(C))   -> Qb [L][64] bf16
//   task 1..4   : K_n = support_n^T @ Wk^T + bk              -> Kb [n][L][64]
//   task 5..8   : V_n = support_n^T @ Wv^T + bv, transposed  -> Vt [n][64][L]
// ---------------------------------------------------------------------------
__global__ __launch_bounds__(256) void proj_kernel(
    const float* __restrict__ query, const float* __restrict__ support,
    const float* __restrict__ Wq, const float* __restrict__ bq,
    const float* __restrict__ Wk, const float* __restrict__ bk,
    const float* __restrict__ Wv, const float* __restrict__ bv,
    unsigned short* __restrict__ Qb, unsigned short* __restrict__ Kb,
    unsigned short* __restrict__ Vt)
{
  __shared__ float xs[64][65];    // xs[c][ll] : input tile (padded)
  __shared__ float Ws[64][65];    // Ws[c][j]  = W[j][c]
  __shared__ float bs[64];

  const int task = blockIdx.x >> 6;          // 0..8
  const int l0   = (blockIdx.x & 63) * 64;
  const int t    = threadIdx.x;

  const float* xin; const float* W; const float* b;
  if (task == 0)      { xin = query;                            W = Wq; b = bq; }
  else if (task <= 4) { xin = support + (task - 1) * (CC * LL); W = Wk; b = bk; }
  else                { xin = support + (task - 5) * (CC * LL); W = Wv; b = bv; }

#pragma unroll
  for (int k = 0; k < 16; ++k) {
    int idx = t + k * 256;
    xs[idx >> 6][idx & 63] = xin[(idx >> 6) * LL + l0 + (idx & 63)];
    Ws[idx & 63][idx >> 6] = W[idx];
  }
  if (t < 64) bs[t] = b[t];
  __syncthreads();

  const int j   = t & 63;
  const int grp = t >> 6;   // 0..3, each handles 16 l's

  float acc[16];
#pragma unroll
  for (int i = 0; i < 16; ++i) acc[i] = bs[j];
  for (int c = 0; c < 64; ++c) {
    float wv = Ws[c][j];
#pragma unroll
    for (int i = 0; i < 16; ++i) acc[i] = fmaf(xs[c][grp * 16 + i], wv, acc[i]);
  }
  if (task == 0) {
#pragma unroll
    for (int i = 0; i < 16; ++i) acc[i] *= 0.125f;   // fold 1/sqrt(C) into Q
  }

  if (task <= 4) {
    unsigned short* o = (task == 0) ? Qb : (Kb + (task - 1) * (LL * CC));
#pragma unroll
    for (int i = 0; i < 16; ++i)
      o[(l0 + grp * 16 + i) * CC + j] = f2bf(acc[i]);
  } else {
    unsigned short* o = Vt + (task - 5) * (CC * LL) + j * LL + l0 + grp * 16;
    unsigned int p[8];
#pragma unroll
    for (int i = 0; i < 8; ++i)
      p[i] = (unsigned)f2bf(acc[2 * i]) | ((unsigned)f2bf(acc[2 * i + 1]) << 16);
    uint4 lo = {p[0], p[1], p[2], p[3]};
    uint4 hi = {p[4], p[5], p[6], p[7]};
    *(uint4*)o = lo;
    *((uint4*)o + 1) = hi;
  }
}

// ---------------------------------------------------------------------------
// Flash attention helpers. Fragment layouts as loaded here (contiguous k for A and B):
//   A (16x32 bf16): lane(m=lane&15, g=lane>>4): row m, k in [16g,16g+16)
//   B (32x16 bf16): lane(n=lane&15, g=lane>>4): col n, k in [16g,16g+16)
//   C (16x16 f32) : lane(n=lane&15, g), vgpr r -> (m = r + 8g, n)
// ---------------------------------------------------------------------------
static __device__ __forceinline__ void load_kblock(const unsigned short* __restrict__ Kn,
                                                   int cb, int lm, int lg,
                                                   v16bf kb[4][2]) {
#pragma unroll
  for (int t = 0; t < 4; ++t) {
    const unsigned short* kr = Kn + (cb + t * 16 + lm) * CC;
    kb[t][0] = load_frag(kr + 16 * lg,      kr + 16 * lg + 8);       // d 0..31
    kb[t][1] = load_frag(kr + 32 + 16 * lg, kr + 32 + 16 * lg + 8);  // d 32..63
  }
}

static __device__ __forceinline__ void attn_step(
    int cb, const unsigned short* __restrict__ Kn, const unsigned short* __restrict__ Vn,
    unsigned short (&Pw)[16][72], int lm, int lg,
    const v16bf& aq0, const v16bf& aq1, const v16bf& vones,
    v16bf cur[4][2], v16bf nxt[4][2],
    v8f o[4], float (&rmax)[8], float (&rsum)[8])
{
  // ---- V fragments for this block: issue first, consumed only after softmax
  v16bf vb[4][2];
#pragma unroll
  for (int t = 0; t < 4; ++t) {
    const unsigned short* vr = Vn + (t * 16 + lm) * LL + cb;
    vb[t][0] = load_frag(vr + 16 * lg,      vr + 16 * lg + 8);       // k 0..31
    vb[t][1] = load_frag(vr + 32 + 16 * lg, vr + 32 + 16 * lg + 8);  // k 32..63
  }

  // distant prefetch hints (global_prefetch_b8)
  if (cb + 128 < LL) {
    __builtin_prefetch(Kn + (cb + 128 + (lm * 2 + lg) * 2) * CC, 0, 0);
    __builtin_prefetch(Vn + (lm * 2 + lg) * 2 * LL + cb + 128, 0, 0);
  }

  // ---- S = Q K^T (scale pre-folded into Q); K was preloaded last step ----
  v8f s[4];
#pragma unroll
  for (int t = 0; t < 4; ++t) {
    v8f acc = (v8f){};
    acc = __builtin_amdgcn_wmma_f32_16x16x32_bf16(false, aq0, false, cur[t][0],
                                                  (short)0, acc, false, false);
    acc = __builtin_amdgcn_wmma_f32_16x16x32_bf16(false, aq1, false, cur[t][1],
                                                  (short)0, acc, false, false);
    s[t] = acc;
  }

  // ---- preload next block's K into the other register buffer ----
  {
    int nc = (cb + 64 < LL) ? (cb + 64) : cb;   // clamped (redundant, harmless)
    load_kblock(Kn, nc, lm, lg, nxt);
  }

  // ---- online softmax: row max via 16-lane-half shuffles (C layout) ----
  float corr[8];
#pragma unroll
  for (int r = 0; r < 8; ++r) {
    float m = fmaxf(fmaxf(s[0][r], s[1][r]), fmaxf(s[2][r], s[3][r]));
    m = fmaxf(m, __shfl_xor(m, 1, 32));
    m = fmaxf(m, __shfl_xor(m, 2, 32));
    m = fmaxf(m, __shfl_xor(m, 4, 32));
    m = fmaxf(m, __shfl_xor(m, 8, 32));
    float nm = fmaxf(rmax[r], m);
    corr[r]  = __expf(rmax[r] - nm);
    rmax[r]  = nm;
    o[0][r] *= corr[r]; o[1][r] *= corr[r]; o[2][r] *= corr[r]; o[3][r] *= corr[r];
  }

  // ---- P = exp(S - rowmax): C layout -> LDS (stride 72) -> A layout ----
#pragma unroll
  for (int t = 0; t < 4; ++t)
#pragma unroll
    for (int r = 0; r < 8; ++r)
      Pw[r + 8 * lg][16 * t + lm] = f2bf(__expf(s[t][r] - rmax[r]));

  const unsigned short* pr = &Pw[lm][0];
  v16bf pa0 = load_frag(pr + 16 * lg,      pr + 16 * lg + 8);   // keys 0..31
  v16bf pa1 = load_frag(pr + 32 + 16 * lg, pr + 32 + 16 * lg + 8);   // keys 32..63

  // ---- row sums on the matrix pipe: z = P · 1 (same C-layout rows as o) ----
  v8f z = (v8f){};
  z = __builtin_amdgcn_wmma_f32_16x16x32_bf16(false, pa0, false, vones,
                                              (short)0, z, false, false);
  z = __builtin_amdgcn_wmma_f32_16x16x32_bf16(false, pa1, false, vones,
                                              (short)0, z, false, false);

  // ---- O += P V : 4 d-tiles x 2 k-halves (V long since in flight) ----
#pragma unroll
  for (int t = 0; t < 4; ++t) {
    o[t] = __builtin_amdgcn_wmma_f32_16x16x32_bf16(false, pa0, false, vb[t][0],
                                                   (short)0, o[t], false, false);
    o[t] = __builtin_amdgcn_wmma_f32_16x16x32_bf16(false, pa1, false, vb[t][1],
                                                   (short)0, o[t], false, false);
  }

#pragma unroll
  for (int r = 0; r < 8; ++r)
    rsum[r] = rsum[r] * corr[r] + z[r];
}

// ---------------------------------------------------------------------------
// 1 block = 4 waves; each wave owns 16 query rows of one support image n.
// ---------------------------------------------------------------------------
__global__ __launch_bounds__(128) void flash_attn_kernel(
    const unsigned short* __restrict__ Qb,   // [L][64]      (pre-scaled)
    const unsigned short* __restrict__ Kb,   // [N][L][64]
    const unsigned short* __restrict__ Vt,   // [N][64][L]   (transposed)
    const float* __restrict__ thrP, const float* __restrict__ tempP,
    float* __restrict__ out)                 // [N][64][L]
{
  __shared__ unsigned short Psh[4][16][72];

  const int w       = threadIdx.x >> 5;
  const int lane    = threadIdx.x & 31;
  const int n       = blockIdx.x >> 6;
  const int rowBase = (blockIdx.x & 63) * 64 + w * 16;
  const int lm      = lane & 15;
  const int lg      = lane >> 4;

  // uniform scalars: issue early
  const float thrRaw = thrP[0];
  const float tmpRaw = tempP[0];

  // Q A-fragments (16 rows, d=64 -> 2 fragments)
  const unsigned short* qr = Qb + (rowBase + lm) * CC;
  const v16bf aq0 = load_frag(qr + 16 * lg,      qr + 16 * lg + 8);
  const v16bf aq1 = load_frag(qr + 32 + 16 * lg, qr + 32 + 16 * lg + 8);
  const v16bf vones = ones_frag();

  v8f o[4];
#pragma unroll
  for (int t = 0; t < 4; ++t) o[t] = (v8f){};
  float rmax[8], rsum[8];
#pragma unroll
  for (int r = 0; r < 8; ++r) { rmax[r] = -3.0e38f; rsum[r] = 0.0f; }

  const unsigned short* Kn = Kb + n * (LL * CC);
  const unsigned short* Vn = Vt + n * (CC * LL);

  // software pipeline: kA/kB double-buffer with static indices (unroll x2)
  v16bf kA[4][2], kB[4][2];
  load_kblock(Kn, 0, lm, lg, kA);

  for (int cb = 0; cb < LL; cb += 128) {
    attn_step(cb,      Kn, Vn, Psh[w], lm, lg, aq0, aq1, vones, kA, kB, o, rmax, rsum);
    attn_step(cb + 64, Kn, Vn, Psh[w], lm, lg, aq0, aq1, vones, kB, kA, o, rmax, rsum);
  }

  // ---- epilogue: attended/Z * sigmoid((1/Z - sigmoid(thr)) * softplus(temp))
  const float thr = 1.0f / (1.0f + __expf(-thrRaw));
  const float tmp = __logf(1.0f + __expf(tmpRaw));
  float fac[8];
#pragma unroll
  for (int r = 0; r < 8; ++r) {
    float inv  = 1.0f / rsum[r];                 // == max_attn
    float mask = 1.0f / (1.0f + __expf(-(inv - thr) * tmp));
    fac[r] = inv * mask;
  }

  // out[n][c][l]; per lane, l = rowBase + 8*lg + r is contiguous -> 2x float4
#pragma unroll
  for (int t = 0; t < 4; ++t) {
    float4 v0 = make_float4(o[t][0] * fac[0], o[t][1] * fac[1],
                            o[t][2] * fac[2], o[t][3] * fac[3]);
    float4 v1 = make_float4(o[t][4] * fac[4], o[t][5] * fac[5],
                            o[t][6] * fac[6], o[t][7] * fac[7]);
    float* op = out + n * (CC * LL) + (t * 16 + lm) * LL + rowBase + 8 * lg;
    *(float4*)op       = v0;
    *((float4*)op + 1) = v1;
  }
}

// ---------------------------------------------------------------------------
extern "C" void kernel_launch(void* const* d_in, const int* in_sizes, int n_in,
                              void* d_out, int out_size, void* d_ws, size_t ws_size,
                              hipStream_t stream) {
  (void)in_sizes; (void)n_in; (void)out_size; (void)ws_size;
  const float* query   = (const float*)d_in[0];
  const float* support = (const float*)d_in[1];
  // d_in[2] = support_labels: computed-but-unused in reference -> omitted
  const float* Wq = (const float*)d_in[3];
  const float* bq = (const float*)d_in[4];
  const float* Wk = (const float*)d_in[5];
  const float* bk = (const float*)d_in[6];
  const float* Wv = (const float*)d_in[7];
  const float* bv = (const float*)d_in[8];
  const float* thr = (const float*)d_in[9];
  const float* tmp = (const float*)d_in[10];

  unsigned short* Qb = (unsigned short*)d_ws;            // L*64 bf16      (0.5 MB)
  unsigned short* Kb = Qb + (size_t)LL * CC;             // N*L*64 bf16    (2 MB)
  unsigned short* Vt = Kb + (size_t)NN * LL * CC;        // N*64*L bf16    (2 MB)

  proj_kernel<<<9 * 64, 256, 0, stream>>>(query, support, Wq, bq, Wk, bk, Wv, bv,
                                          Qb, Kb, Vt);
  flash_attn_kernel<<<NN * (LL / 64), 128, 0, stream>>>(Qb, Kb, Vt, thr, tmp,
                                                        (float*)d_out);
}
